// Attention_71459665871434
// MI455X (gfx1250) — compile-verified, matrix operand order checked by reading, not run on hardware
//
#include <hip/hip_runtime.h>
#include <hip/hip_bf16.h>
#include <math.h>

// Problem constants (match reference)
#define BB 32
#define TT 4096
#define DD 256
#define AA 128

typedef __attribute__((ext_vector_type(16))) __bf16 v16bf;
typedef __attribute__((ext_vector_type(8)))  float  v8f;

// ---------------------------------------------------------------------------
// Workspace layout:
//   [0, 65536)            : W_w converted to bf16, pre-swizzled into the exact
//                           v_wmma_f32_16x16x32_bf16 B-fragment layout:
//                           fragment (ntile, kstep) -> 32 lanes x 16 bf16.
//   [65536, 65536+512KB)  : score / alpha, B*T floats.
// ---------------------------------------------------------------------------
#define WSWZ_BYTES 65536
#define N_TILES 8   // A / 16
#define K_STEPS 8   // D / 32

// Hardware tanh on gfx1250 (V_TANH_F32, TRANS32 pipe). Guarded so it compiles
// on toolchains that name the builtin differently; fallback is branchless
// tanh(x) = 1 - 2/(exp(2x)+1) using v_exp_f32 + v_rcp_f32 (no EXEC juggling,
// saturates correctly at +-1 via inf/0 behavior of exp).
__device__ inline float hw_tanh(float x) {
#if __has_builtin(__builtin_amdgcn_tanhf)
    return __builtin_amdgcn_tanhf(x);
#elif __has_builtin(__builtin_amdgcn_tanh_f32)
    return __builtin_amdgcn_tanh_f32(x);
#else
    float t = __expf(2.0f * x);
    return 1.0f - 2.0f * __builtin_amdgcn_rcpf(t + 1.0f);
#endif
}

// --- Kernel 0: convert + swizzle W_w (f32 [A,D]) into bf16 B-fragments ------
// ISA 16-bit fragment layout: lane L in [0,16): row = L, elems e:
//   K = e + 8*half + 8*(e>>3)   (half = lane>>4)
__global__ void attnpool_wconv(const float* __restrict__ W, __bf16* __restrict__ out) {
    int idx = blockIdx.x * blockDim.x + threadIdx.x;   // one (n,k,lane) each
    if (idx >= N_TILES * K_STEPS * 32) return;
    int lane = idx & 31;
    int k    = (idx >> 5) & 7;
    int n    = idx >> 8;
    int m    = lane & 15;
    int half = lane >> 4;
    const float* row = W + (n * 16 + m) * DD + k * 32;  // W_w row-major [A,D]
    __bf16* o = out + (size_t)idx * 16;
#pragma unroll
    for (int e = 0; e < 16; ++e) {
        int kk = e + 8 * half + 8 * (e >> 3);
        o[e] = (__bf16)row[kk];
    }
}

__device__ inline void pack8(v16bf& v, int off, float4 a, float4 b) {
    v[off + 0] = (__bf16)a.x; v[off + 1] = (__bf16)a.y;
    v[off + 2] = (__bf16)a.z; v[off + 3] = (__bf16)a.w;
    v[off + 4] = (__bf16)b.x; v[off + 5] = (__bf16)b.y;
    v[off + 6] = (__bf16)b.z; v[off + 7] = (__bf16)b.w;
}

// --- Kernel 1: fused score GEMM: score = u_w . tanh(h @ W^T + b_w) ----------
// 4 waves / block; each wave owns a 16-row tile of the flattened [B*T] axis.
__global__ __launch_bounds__(128) void attnpool_score(
    const float*  __restrict__ h,      // [B*T, D] f32
    const v16bf*  __restrict__ wfrag,  // swizzled bf16 W fragments
    const float*  __restrict__ b_w,    // [A]
    const float*  __restrict__ u_w,    // [A]
    float*        __restrict__ score)  // [B*T]
{
    const int wave = threadIdx.x >> 5;
    const int lane = threadIdx.x & 31;
    const int m    = lane & 15;
    const int half = lane >> 4;

    const long tile = (long)blockIdx.x * 4 + wave;  // 8192 tiles total, exact
    const long row0 = tile * 16;
    const float* hrow = h + (row0 + m) * DD;

    // Load this wave's 16x256 A-panel once, converting f32 -> bf16 in the
    // exact A-fragment layout (lane half selects K sub-blocks of 8).
    v16bf a[K_STEPS];
#pragma unroll
    for (int k = 0; k < K_STEPS; ++k) {
        const float4* p1 = (const float4*)(hrow + k * 32 +      8 * half);
        const float4* p2 = (const float4*)(hrow + k * 32 + 16 + 8 * half);
        pack8(a[k], 0, p1[0], p1[1]);
        pack8(a[k], 8, p2[0], p2[1]);
    }

    float part[8];
#pragma unroll
    for (int r = 0; r < 8; ++r) part[r] = 0.0f;

    // Sweep the 8 N-tiles of the A=128 attention dim; fuse bias+tanh+dot.
    for (int n = 0; n < N_TILES; ++n) {
        v8f c = {};
#pragma unroll
        for (int k = 0; k < K_STEPS; ++k) {
            v16bf b = wfrag[(n * K_STEPS + k) * 32 + lane];
            c = __builtin_amdgcn_wmma_f32_16x16x32_bf16(
                    false, a[k], false, b, (short)0, c, false, false);
        }
        const int nidx = n * 16 + m;            // C/D layout: N = lane&15
        const float bw = b_w[nidx];
        const float uw = u_w[nidx];
#pragma unroll
        for (int r = 0; r < 8; ++r)             // c[r]: M = r + 8*half
            part[r] += hw_tanh(c[r] + bw) * uw;
    }

    // Reduce over N (16 lanes within each half-wave); lanes 0 and 16 write.
#pragma unroll
    for (int r = 0; r < 8; ++r) {
        float v = part[r];
        v += __shfl_xor(v, 1, 32);
        v += __shfl_xor(v, 2, 32);
        v += __shfl_xor(v, 4, 32);
        v += __shfl_xor(v, 8, 32);
        if (m == 0) score[row0 + 8 * half + r] = v;
    }
}

// --- Kernel 2: softmax over T per batch (in place: score -> alpha) ----------
__global__ __launch_bounds__(256) void attnpool_softmax(float* __restrict__ score) {
    const int b = blockIdx.x;
    float* s = score + (long)b * TT;
    __shared__ float red[256];
    const int tid = threadIdx.x;

    float mx = -INFINITY;
    for (int t = tid; t < TT; t += 256) mx = fmaxf(mx, s[t]);
    red[tid] = mx; __syncthreads();
    for (int w = 128; w > 0; w >>= 1) {
        if (tid < w) red[tid] = fmaxf(red[tid], red[tid + w]);
        __syncthreads();
    }
    mx = red[0]; __syncthreads();

    float sum = 0.0f;
    for (int t = tid; t < TT; t += 256) {
        float e = __expf(s[t] - mx);
        s[t] = e;
        sum += e;
    }
    red[tid] = sum; __syncthreads();
    for (int w = 128; w > 0; w >>= 1) {
        if (tid < w) red[tid] += red[tid + w];
        __syncthreads();
    }
    const float inv = 1.0f / red[0];
    for (int t = tid; t < TT; t += 256) s[t] *= inv;
}

// --- Kernel 3: pooling s[b,d] = sum_t alpha[b,t] * h[b,t,d] -----------------
// grid (B, 16): T split 16 ways for occupancy; h mostly hits L2 on this pass.
__global__ __launch_bounds__(256) void attnpool_pool(
    const float* __restrict__ h, const float* __restrict__ alpha,
    float* __restrict__ out)
{
    const int b  = blockIdx.x;
    const int d  = threadIdx.x;                 // D == blockDim == 256
    const int t0 = blockIdx.y * (TT / 16);
    const float* hb = h + ((long)b * TT + t0) * DD;
    const float* ab = alpha + (long)b * TT + t0;
    float acc = 0.0f;
#pragma unroll 4
    for (int t = 0; t < TT / 16; ++t)
        acc += ab[t] * hb[(long)t * DD + d];
    atomicAdd(&out[b * DD + d], acc);
}

extern "C" void kernel_launch(void* const* d_in, const int* in_sizes, int n_in,
                              void* d_out, int out_size, void* d_ws, size_t ws_size,
                              hipStream_t stream) {
    const float* h   = (const float*)d_in[0];   // [B,T,D]
    const float* W_w = (const float*)d_in[1];   // [A,D]
    const float* b_w = (const float*)d_in[2];   // [A]
    const float* u_w = (const float*)d_in[3];   // [A]
    float* out = (float*)d_out;                 // [B,D]

    char*  wsb   = (char*)d_ws;
    __bf16* wswz = (__bf16*)wsb;
    float* score = (float*)(wsb + WSWZ_BYTES);

    // d_out accumulated via atomics -> zero it first (graph-capturable).
    hipMemsetAsync(d_out, 0, (size_t)BB * DD * sizeof(float), stream);

    attnpool_wconv<<<8, 256, 0, stream>>>(W_w, wswz);

    // B*T / (16 rows * 4 waves) = 131072/64 = 2048 blocks
    attnpool_score<<<2048, 128, 0, stream>>>(h, (const v16bf*)wswz, b_w, u_w, score);

    attnpool_softmax<<<BB, 256, 0, stream>>>(score);

    attnpool_pool<<<dim3(BB, 16), 256, 0, stream>>>(h, score, out);
}
